// LatentNeuralSDE_73744588472731
// MI455X (gfx1250) — matrix operand order repaired and checked by reading, not compiled, not run on hardware
//
#include <hip/hip_runtime.h>
#include <cstdint>
#include <cstddef>

typedef __attribute__((ext_vector_type(16))) _Float16 v16h;
typedef __attribute__((ext_vector_type(8)))  _Float16 v8h;
typedef __attribute__((ext_vector_type(8)))  float    v8f;

#define LAT 10
#define NW  8                      // waves per block (wave32)
#define ROWS_PER_BLOCK (NW * 16)   // 128 rows / block
#define XS  136                    // X / H row stride in halfs (128 + 8 pad -> conflict-free)
#define GS  40                     // diffusion buffers row stride in halfs (32 + 8 pad)

// ---- LDS layout (byte offsets) ----
#define OFF_W1   0u        // dW1 frags: 4 ktile x 8 ntile x 512 halfs = 32768 B
#define OFF_W2   32768u    // dW2 frags: 32768 B
#define OFF_W3   65536u    // dW3 frags: 4 x 512 halfs = 4096 B
#define OFF_G1   69632u    // gW1 frags: 2 x 512 halfs = 2048 B
#define OFF_G2   71680u    // gW2 frags: 1 x 512 halfs = 1024 B
#define OFF_B2   73216u    // db2: 128 f32
#define OFF_B3   73728u    // db3: 16 f32 (padded)
#define OFF_GB2  73920u    // gb2: 16 f32 (padded)
#define OFF_X    73984u    // layer-1 input, f16: 128 rows x XS = 34816 B
#define OFF_H1   108800u   // hidden 1, f16: 34816 B
#define OFF_H2   143616u   // hidden 2, f16: 34816 B
#define OFF_XG   178432u   // diffusion input, f16: 128 x GS = 10240 B
#define OFF_HG   188672u   // diffusion hidden, f16: 10240 B
#define OFF_Z    198912u   // f32 state: 128 rows x 16 = 8192 B
#define SMEM_TOTAL 207104u

__device__ __forceinline__ v8f wmma16(v16h a, v16h b, v8f c) {
    // v_wmma_f32_16x16x32_f16
    return __builtin_amdgcn_wmma_f32_16x16x32_f16(false, a, false, b, (short)0, c, false, false);
}

// ---- branch-free activations ----
// tanh: prefer the CDNA5 V_TANH_F32 transcendental.
#if __has_builtin(__builtin_amdgcn_tanhf)
__device__ __forceinline__ float act_tanh(float x) { return __builtin_amdgcn_tanhf(x); }
#elif __has_builtin(__builtin_amdgcn_tanh_f32)
__device__ __forceinline__ float act_tanh(float x) { return __builtin_amdgcn_tanh_f32(x); }
#else
__device__ __forceinline__ float act_tanh(float x) {
    // ISA lists TANH among gfx1250 TRANS32 ops; no clang builtin on this
    // toolchain, so emit it directly. v_nop satisfies the TRANS-result
    // hazard (1 op before the output may be consumed) since the compiler
    // cannot see through the asm.
    float r;
    asm("v_tanh_f32 %0, %1\n\tv_nop" : "=v"(r) : "v"(x));
    return r;
}
#endif

__device__ __forceinline__ float act_softplus(float x) {
    // log1p(exp(x)) = max(x,0) + log(1 + exp(-|x|)).
    // Raw v_exp_f32 / v_log_f32 (log2/exp2) with explicit scaling: the log
    // argument is in (1,2], never denormal, so no libm range fixups needed.
    const float LOG2E = 1.4426950408889634f;
    const float LN2   = 0.6931471805599453f;
    float e = __builtin_amdgcn_exp2f(-fabsf(x) * LOG2E);
    return fmaxf(x, 0.0f) + LN2 * __builtin_amdgcn_logf(1.0f + e);
}

// A-fragment load (16x32 f16, row-major source in LDS).
// lanes 0-15: M=lane, K = {kt..kt+7, kt+16..kt+23}
// lanes 16-31: M=lane-16, K = {kt+8..kt+15, kt+24..kt+31}
__device__ __forceinline__ v16h loadA(const _Float16* base, int stride_h, int lane, int ktile_off) {
    const int m    = lane & 15;
    const int koff = ktile_off + ((lane >> 4) << 3);
    const _Float16* p = base + m * stride_h + koff;
    v8h lo = *(const v8h*)(p);
    v8h hi = *(const v8h*)(p + 16);
    v16h a;
#pragma unroll
    for (int i = 0; i < 8; ++i) { a[i] = lo[i]; a[i + 8] = hi[i]; }
    return a;
}

// B-fragment load from pre-swizzled LDS: tile = kt*NT + nt, lane gets 16 contiguous halfs.
__device__ __forceinline__ v16h loadB(const _Float16* frag, int tile, int lane) {
    return *(const v16h*)(frag + tile * 512 + lane * 16);
}

// Swizzle a row-major [Kreal x Nreal] f32 weight into WMMA B-fragment order (f16)
// in LDS. Fragment mapping: k = kt*32 + (lane>>4)*8 + (i&7) + (i>>3)*16, n = nt*16 + (lane&15).
// Optionally folds a bias vector into padded-K row `bias_k` (the matching input
// column is set to 1.0 at init, so the WMMA adds the bias for free).
__device__ void load_frag(const float* W, int Kreal, int Nreal, int KT, int NT,
                          _Float16* dst, int tid, int nth,
                          const float* bias = nullptr, int bias_k = -1) {
    const int total = KT * NT * 512;
    for (int e = tid; e < total; e += nth) {
        int tile = e >> 9;
        int r    = e & 511;
        int lane = r >> 4;
        int i    = r & 15;
        int kt = tile / NT, nt = tile % NT;
        int k = kt * 32 + ((lane >> 4) << 3) + (i & 7) + ((i >> 3) << 4);
        int n = nt * 16 + (lane & 15);
        float v = 0.0f;
        if (n < Nreal) {
            if (k < Kreal)          v = W[k * Nreal + n];
            else if (k == bias_k)   v = bias[n];
        }
        dst[e] = (_Float16)v;
    }
}

__global__ __launch_bounds__(256, 1) void sde_kernel(
    const float* __restrict__ z0, const float* __restrict__ activity,
    const float* __restrict__ rest, const float* __restrict__ ts,
    const float* __restrict__ noise,
    const float* __restrict__ aW1, const float* __restrict__ ab1,
    const float* __restrict__ aW2, const float* __restrict__ ab2,
    const float* __restrict__ rW1, const float* __restrict__ rb1,
    const float* __restrict__ rW2, const float* __restrict__ rb2,
    const float* __restrict__ dW1, const float* __restrict__ db1,
    const float* __restrict__ dW2, const float* __restrict__ db2,
    const float* __restrict__ dW3, const float* __restrict__ db3,
    const float* __restrict__ gW1, const float* __restrict__ gb1,
    const float* __restrict__ gW2, const float* __restrict__ gb2,
    float* __restrict__ out, int Bt, int nstep)
{
    extern __shared__ char smem[];
    _Float16* W1f = (_Float16*)(smem + OFF_W1);
    _Float16* W2f = (_Float16*)(smem + OFF_W2);
    _Float16* W3f = (_Float16*)(smem + OFF_W3);
    _Float16* G1f = (_Float16*)(smem + OFF_G1);
    _Float16* G2f = (_Float16*)(smem + OFF_G2);
    float*    B2  = (float*)(smem + OFF_B2);
    float*    B3  = (float*)(smem + OFF_B3);
    float*    GB2 = (float*)(smem + OFF_GB2);
    _Float16* Xb  = (_Float16*)(smem + OFF_X);
    _Float16* H1  = (_Float16*)(smem + OFF_H1);
    _Float16* H2  = (_Float16*)(smem + OFF_H2);
    _Float16* XG  = (_Float16*)(smem + OFF_XG);
    _Float16* HG  = (_Float16*)(smem + OFF_HG);
    float*    Zb  = (float*)(smem + OFF_Z);

    const int tid = threadIdx.x;
    const int nth = blockDim.x;
    const int rowbase = blockIdx.x * ROWS_PER_BLOCK;

    // ---- Stage weights into LDS in WMMA B-fragment order (f16) ----
    load_frag(dW1, 107, 128, 4, 8, W1f, tid, nth, db1, 107); // bias folded at k=107
    load_frag(dW2, 128, 128, 4, 8, W2f, tid, nth);
    load_frag(dW3, 128,  10, 4, 1, W3f, tid, nth);           // N padded 10->16
    load_frag(gW1,  11,  32, 1, 2, G1f, tid, nth, gb1, 11);  // bias folded at k=11
    load_frag(gW2,  32,  10, 1, 1, G2f, tid, nth);
    for (int i = tid; i < 128; i += nth) { B2[i] = db2[i]; }
    for (int i = tid; i < 16;  i += nth) {
        B3[i]  = (i < LAT) ? db3[i] : 0.0f;
        GB2[i] = (i < LAT) ? gb2[i] : 0.0f;
    }

    // ---- Context encoders (once per row, scalar VALU) + state init ----
    if (tid < ROWS_PER_BLOCK) {
        const int r   = tid;            // local row
        const int row = rowbase + r;
        _Float16* Xr  = Xb + r * XS;
        _Float16* XGr = XG + r * GS;
        float*    Zr  = Zb + r * 16;

        float ha[32];
#pragma unroll
        for (int j = 0; j < 32; ++j) {
            float s = ab1[j];
#pragma unroll
            for (int i = 0; i < 6; ++i) s += activity[row * 6 + i] * aW1[i * 32 + j];
            ha[j] = fmaxf(s, 0.0f);
        }
        for (int j = 0; j < 64; ++j) {
            float s = ab2[j];
#pragma unroll
            for (int i = 0; i < 32; ++i) s += ha[i] * aW2[i * 64 + j];
            Xr[10 + j] = (_Float16)s;
        }
        float hr[16];
#pragma unroll
        for (int j = 0; j < 16; ++j) {
            float s = rb1[j];
#pragma unroll
            for (int i = 0; i < 3; ++i) s += rest[row * 3 + i] * rW1[i * 16 + j];
            hr[j] = fmaxf(s, 0.0f);
        }
        for (int j = 0; j < 32; ++j) {
            float s = rb2[j];
#pragma unroll
            for (int i = 0; i < 16; ++i) s += hr[i] * rW2[i * 32 + j];
            Xr[74 + j] = (_Float16)s;
        }
#pragma unroll
        for (int n = 0; n < LAT; ++n) {
            float zv = z0[row * LAT + n];
            Zr[n] = zv;
            Xr[n]  = (_Float16)zv;
            XGr[n] = (_Float16)zv;
            out[row * LAT + n] = zv;           // out[0] = z0
        }
#pragma unroll
        for (int n = LAT; n < 16; ++n) Zr[n] = 0.0f;
        Xr[107] = (_Float16)1.0f;                                // bias-ones column
        for (int c = 108; c < XS; ++c) Xr[c] = (_Float16)0.0f;   // zero K-pad + row pad
        XGr[11] = (_Float16)1.0f;                                // bias-ones column
        for (int c = 12; c < GS; ++c) XGr[c] = (_Float16)0.0f;   // zero diffusion K-pad
    }
    __syncthreads();

    // ---- Main SDE loop: each wave owns 16 rows; no cross-wave deps ----
    const int w    = tid >> 5;
    const int lane = tid & 31;
    _Float16* Xw  = Xb + w * 16 * XS;
    _Float16* H1w = H1 + w * 16 * XS;
    _Float16* H2w = H2 + w * 16 * XS;
    _Float16* XGw = XG + w * 16 * GS;
    _Float16* HGw = HG + w * 16 * GS;
    float*    Zw  = Zb + w * 16 * 16;
    const int nD  = lane & 15;          // D-matrix column for this lane
    const int mD0 = (lane >> 4) << 3;   // first D-matrix row for this lane
    const int wave_rowbase = rowbase + w * 16;

    for (int s = 0; s < nstep; ++s) {
        float t   = ts[s];
        float dt  = ts[s + 1] - t;
        float sdt = sqrtf(dt);

        // Issue this step's noise loads up front; consumed at the end of the
        // step, so HBM latency is hidden behind the whole step's compute.
        float eps[8];
        if (nD < LAT) {
#pragma unroll
            for (int i = 0; i < 8; ++i) {
                const int row = wave_rowbase + mD0 + i;
                eps[i] = noise[((size_t)s * (size_t)Bt + (size_t)row) * LAT + nD];
            }
        }

        // refresh the t column (lanes 0..15 cover the 16 rows)
        if (lane < 16) {
            Xw[lane * XS + 106] = (_Float16)t;
            XGw[lane * GS + 10] = (_Float16)t;
        }

        // ---- diffusion net first (independent of drift MLP) ----
        // layer 1: [16x32(pad)] x [32x32] (bias folded), softplus
        {
            v16h Ag = loadA(XGw, GS, lane, 0);
#pragma unroll
            for (int nt = 0; nt < 2; ++nt) {
                const int ncol = nt * 16 + nD;
                v8f c;
#pragma unroll
                for (int i = 0; i < 8; ++i) c[i] = 0.0f;
                c = wmma16(Ag, loadB(G1f, nt, lane), c);
#pragma unroll
                for (int i = 0; i < 8; ++i)
                    HGw[(mD0 + i) * GS + ncol] = (_Float16)act_softplus(c[i]);
            }
        }
        // layer 2: [16x32] x [32x16(pad)] -> diffusion in regs
        v8f cg;
        {
            float bv = GB2[nD];
#pragma unroll
            for (int i = 0; i < 8; ++i) cg[i] = bv;
            cg = wmma16(loadA(HGw, GS, lane, 0), loadB(G2f, 0, lane), cg);
        }

        // ---- drift layer 1: [16x128] x [128x128] (bias folded), tanh ----
        {
            v16h A0 = loadA(Xw, XS, lane, 0);
            v16h A1 = loadA(Xw, XS, lane, 32);
            v16h A2 = loadA(Xw, XS, lane, 64);
            v16h A3 = loadA(Xw, XS, lane, 96);
#pragma unroll
            for (int nt = 0; nt < 8; ++nt) {
                const int ncol = nt * 16 + nD;
                v8f c;
#pragma unroll
                for (int i = 0; i < 8; ++i) c[i] = 0.0f;
                c = wmma16(A0, loadB(W1f, 0 * 8 + nt, lane), c);
                c = wmma16(A1, loadB(W1f, 1 * 8 + nt, lane), c);
                c = wmma16(A2, loadB(W1f, 2 * 8 + nt, lane), c);
                c = wmma16(A3, loadB(W1f, 3 * 8 + nt, lane), c);
#pragma unroll
                for (int i = 0; i < 8; ++i)
                    H1w[(mD0 + i) * XS + ncol] = (_Float16)act_tanh(c[i]);
            }
        }
        // ---- drift layer 2: [16x128] x [128x128], tanh ----
        {
            v16h A0 = loadA(H1w, XS, lane, 0);
            v16h A1 = loadA(H1w, XS, lane, 32);
            v16h A2 = loadA(H1w, XS, lane, 64);
            v16h A3 = loadA(H1w, XS, lane, 96);
#pragma unroll
            for (int nt = 0; nt < 8; ++nt) {
                const int ncol = nt * 16 + nD;
                float bv = B2[ncol];
                v8f c;
#pragma unroll
                for (int i = 0; i < 8; ++i) c[i] = bv;
                c = wmma16(A0, loadB(W2f, 0 * 8 + nt, lane), c);
                c = wmma16(A1, loadB(W2f, 1 * 8 + nt, lane), c);
                c = wmma16(A2, loadB(W2f, 2 * 8 + nt, lane), c);
                c = wmma16(A3, loadB(W2f, 3 * 8 + nt, lane), c);
#pragma unroll
                for (int i = 0; i < 8; ++i)
                    H2w[(mD0 + i) * XS + ncol] = (_Float16)act_tanh(c[i]);
            }
        }
        // ---- drift layer 3: [16x128] x [128x16(pad)] -> drift in regs ----
        v8f c3;
        {
            float bv = B3[nD];
#pragma unroll
            for (int i = 0; i < 8; ++i) c3[i] = bv;
            c3 = wmma16(loadA(H2w, XS, lane, 0),  loadB(W3f, 0, lane), c3);
            c3 = wmma16(loadA(H2w, XS, lane, 32), loadB(W3f, 1, lane), c3);
            c3 = wmma16(loadA(H2w, XS, lane, 64), loadB(W3f, 2, lane), c3);
            c3 = wmma16(loadA(H2w, XS, lane, 96), loadB(W3f, 3, lane), c3);
        }

        // ---- Euler-Maruyama update; keep z in f32 LDS, emit output ----
        if (nD < LAT) {
#pragma unroll
            for (int i = 0; i < 8; ++i) {
                const int m2  = mD0 + i;
                const int row = wave_rowbase + m2;
                float zn  = Zw[m2 * 16 + nD] + c3[i] * dt + cg[i] * sdt * eps[i];
                Zw[m2 * 16 + nD] = zn;
                Xw[m2 * XS + nD]  = (_Float16)zn;
                XGw[m2 * GS + nD] = (_Float16)zn;
                out[((size_t)(s + 1) * (size_t)Bt + (size_t)row) * LAT + nD] = zn;
            }
        }
    }
}

extern "C" void kernel_launch(void* const* d_in, const int* in_sizes, int n_in,
                              void* d_out, int out_size, void* d_ws, size_t ws_size,
                              hipStream_t stream) {
    const float* z0       = (const float*)d_in[0];
    const float* activity = (const float*)d_in[1];
    const float* rest     = (const float*)d_in[2];
    const float* ts       = (const float*)d_in[3];
    const float* noise    = (const float*)d_in[4];
    const float* aW1 = (const float*)d_in[5];
    const float* ab1 = (const float*)d_in[6];
    const float* aW2 = (const float*)d_in[7];
    const float* ab2 = (const float*)d_in[8];
    const float* rW1 = (const float*)d_in[9];
    const float* rb1 = (const float*)d_in[10];
    const float* rW2 = (const float*)d_in[11];
    const float* rb2 = (const float*)d_in[12];
    const float* dW1 = (const float*)d_in[13];
    const float* db1 = (const float*)d_in[14];
    const float* dW2 = (const float*)d_in[15];
    const float* db2 = (const float*)d_in[16];
    const float* dW3 = (const float*)d_in[17];
    const float* db3 = (const float*)d_in[18];
    const float* gW1 = (const float*)d_in[19];
    const float* gb1 = (const float*)d_in[20];
    const float* gW2 = (const float*)d_in[21];
    const float* gb2 = (const float*)d_in[22];
    float* out = (float*)d_out;

    const int Bt    = in_sizes[0] / LAT;   // 32768
    const int nstep = in_sizes[3] - 1;     // 100
    const int grid  = Bt / ROWS_PER_BLOCK; // 256

    (void)hipFuncSetAttribute((const void*)sde_kernel,
                              hipFuncAttributeMaxDynamicSharedMemorySize,
                              (int)SMEM_TOTAL);

    sde_kernel<<<grid, 256, SMEM_TOTAL, stream>>>(
        z0, activity, rest, ts, noise,
        aW1, ab1, aW2, ab2, rW1, rb1, rW2, rb2,
        dW1, db1, dW2, db2, dW3, db3,
        gW1, gb1, gW2, gb2,
        out, Bt, nstep);
}
